// LocalInfoNCELoss_5437428597449
// MI455X (gfx1250) — matrix operand order repaired and checked by reading, not compiled, not run on hardware
//
#include <hip/hip_runtime.h>
#include <hip/hip_bf16.h>
#include <stdint.h>

// ---------------------------------------------------------------------------
// LocalInfoNCE loss, MI455X (gfx1250, wave32, WMMA bf16 16x16x32)
//   N=32 batches, L=1024 grid pts, C=512 channels, M=4096 negatives
// Register-blocked WMMA GEMMs + fused logsumexp epilogues.
// ---------------------------------------------------------------------------

#define N_BATCH 32
#define L_GRID  1024
#define C_DIM   512
#define M_NEG   4096
#define ROWS    (N_BATCH * L_GRID)   // 32768

typedef __attribute__((ext_vector_type(16))) __bf16 v16bf;
typedef __attribute__((ext_vector_type(8)))  float  v8f;

struct alignas(16) Pair128 { uint4 lo, hi; };

static __device__ __forceinline__ float bf2f(unsigned short u) {
  unsigned v = (unsigned)u << 16;
  return __builtin_bit_cast(float, v);
}
static __device__ __forceinline__ unsigned short f2bf(float f) {
  unsigned u = __builtin_bit_cast(unsigned, f);
  u += 0x7fffu + ((u >> 16) & 1u);   // round-to-nearest-even
  return (unsigned short)(u >> 16);
}

// XOR-butterfly sum over each 16-lane half via ds_swizzle_b32
// (group-of-32 mode: offset = xor_mask<<10 | and_mask 0x1f). Masks 1/2/4/8
// never cross the lane-16 boundary, matching the WMMA C/D N-striping.
static __device__ __forceinline__ float red16(float s) {
  s += __builtin_bit_cast(float, __builtin_amdgcn_ds_swizzle(__builtin_bit_cast(int, s), 0x041f));
  s += __builtin_bit_cast(float, __builtin_amdgcn_ds_swizzle(__builtin_bit_cast(int, s), 0x081f));
  s += __builtin_bit_cast(float, __builtin_amdgcn_ds_swizzle(__builtin_bit_cast(int, s), 0x101f));
  s += __builtin_bit_cast(float, __builtin_amdgcn_ds_swizzle(__builtin_bit_cast(int, s), 0x201f));
  return s;
}

// A fragment (16x32 bf16, MxK): lane row = lane&15, K offset 8*(lane>>4)
// folded into p by caller; elements = {p[0..7], p[16..23]}.
static __device__ __forceinline__ v16bf load_frag_a(const unsigned short* p) {
  Pair128 t;
  t.lo = *(const uint4*)(p);
  t.hi = *(const uint4*)(p + 16);
  return __builtin_bit_cast(v16bf, t);
}
// B fragment (32x16 bf16, KxN): lane col = lane&15, K offset 8*(lane>>4)
// folded into p; elements = {p[0..7], p[16..23]}.
static __device__ __forceinline__ v16bf load_frag_b(const unsigned short* p) {
  Pair128 t;
  t.lo = *(const uint4*)(p);
  t.hi = *(const uint4*)(p + 16);
  return __builtin_bit_cast(v16bf, t);
}

static __device__ __forceinline__ v8f wmma_bf16(v16bf a, v16bf b, v8f c) {
  return __builtin_amdgcn_wmma_f32_16x16x32_bf16(false, a, false, b,
                                                 (short)0, c, false, false);
}

// ---------------------------------------------------------------------------
__global__ void k_zero(float* __restrict__ p, int n) {
  int i = blockIdx.x * blockDim.x + threadIdx.x;
  if (i < n) p[i] = 0.f;
}

__global__ void k_cvt_w(const float* __restrict__ w, unsigned short* __restrict__ wbf) {
  int i = (blockIdx.x * blockDim.x + threadIdx.x) * 4;
#pragma unroll
  for (int j = 0; j < 4; ++j) wbf[i + j] = f2bf(w[i + j]);
}

// One wave per row: normalize over C=512, emit bf16. 8 waves / block.
__global__ void k_normalize(const float* __restrict__ emb, unsigned short* __restrict__ ebf) {
  const int lane = threadIdx.x & 31;
  const int row  = blockIdx.x * (blockDim.x >> 5) + (threadIdx.x >> 5);
  const float* p = emb + (size_t)row * C_DIM + lane * 16;
  float v[16];
  float ss = 0.f;
#pragma unroll
  for (int i = 0; i < 16; ++i) { v[i] = p[i]; ss += v[i] * v[i]; }
#pragma unroll
  for (int m = 16; m >= 1; m >>= 1) ss += __shfl_xor(ss, m, 32);
  const float scale = 1.f / fmaxf(sqrtf(ss), 1e-12f);
  unsigned int out[8];
#pragma unroll
  for (int i = 0; i < 8; ++i)
    out[i] = (unsigned)f2bf(v[2 * i] * scale) | ((unsigned)f2bf(v[2 * i + 1] * scale) << 16);
  uint4* q = (uint4*)(ebf + (size_t)row * C_DIM + lane * 16);
  q[0] = make_uint4(out[0], out[1], out[2], out[3]);
  q[1] = make_uint4(out[4], out[5], out[6], out[7]);
}

// One wave per negative row: gather 512 bf16 (1 KiB) per index.
__global__ void k_gather(const unsigned short* __restrict__ ebf,
                         const int* __restrict__ idx,
                         unsigned short* __restrict__ neg) {
  const int lane = threadIdx.x & 31;
  const int m = blockIdx.x * (blockDim.x >> 5) + (threadIdx.x >> 5);
  const int r = idx[m];
  const uint4* s = (const uint4*)(ebf + (size_t)r * C_DIM);
  uint4*       d = (uint4*)(neg + (size_t)m * C_DIM);
#pragma unroll
  for (int i = 0; i < 2; ++i) d[lane + 32 * i] = s[lane + 32 * i];
}

// target[n] = W @ e[n] fused with sim_pos = rowdot(e, target).
// Each wave: 64(k) x 16(c) output block = 4 accumulators; the LDS-staged
// B fragment (transposed e-slice) is reused 4x. Per-wave LDS slice +
// wave_barrier (LDS is HW-in-order within a wave; no workgroup barrier).
__global__ void __launch_bounds__(128)
k_target_simpos(const unsigned short* __restrict__ Wbf,
                const unsigned short* __restrict__ Ebf,
                float* __restrict__ simpos) {
  __shared__ unsigned short smem[4][16 * 32];   // per-wave [c][l] slice, 1 KiB
  const int lane = threadIdx.x & 31;
  const int wid  = threadIdx.x >> 5;
  const int tile = blockIdx.x * 4 + wid;        // 32(c) * 16(kblk) * 32(n) = 16384
  const int c0 = (tile & 31) << 4;
  const int k0 = ((tile >> 5) & 15) << 6;       // 64-row k block
  const int n  = tile >> 9;
  const int lr = lane & 15;
  const int lh = lane >> 4;
  unsigned short* sm = smem[wid];

  const unsigned short* Abase = Wbf + (size_t)(k0 + lr) * L_GRID + 8 * lh;
  const unsigned short* Eb    = Ebf + (size_t)n * L_GRID * C_DIM;

  v8f acc[4] = {};
  for (int l0 = 0; l0 < L_GRID; l0 += 32) {
    // stage rows l0..l0+31, cols c0..c0+15 of e[n], transposed to [c][l]
    const unsigned int* gp = (const unsigned int*)(Eb + (size_t)(l0 + lane) * C_DIM + c0);
    unsigned int wv[8];
#pragma unroll
    for (int i = 0; i < 8; ++i) wv[i] = gp[i];
#pragma unroll
    for (int i = 0; i < 8; ++i) {
      sm[(2 * i    ) * 32 + lane] = (unsigned short)(wv[i] & 0xffffu);
      sm[(2 * i + 1) * 32 + lane] = (unsigned short)(wv[i] >> 16);
    }
    __builtin_amdgcn_wave_barrier();            // LDS in-order per wave
    v16bf b = load_frag_b(sm + lr * 32 + 8 * lh);
#pragma unroll
    for (int t = 0; t < 4; ++t) {
      v16bf a = load_frag_a(Abase + (size_t)t * 16 * L_GRID + l0);
      acc[t] = wmma_bf16(a, b, acc[t]);
    }
    __builtin_amdgcn_wave_barrier();            // before next overwrite
  }
  // epilogue: sim_pos[n,k] += sum_c e[n,k,c] * target[n,k,c]
#pragma unroll
  for (int t = 0; t < 4; ++t) {
#pragma unroll
    for (int j = 0; j < 8; ++j) {
      const int row = k0 + 16 * t + j + 8 * lh;  // D: VGPR j -> M=j (+8 high half)
      float s = acc[t][j] * bf2f(Ebf[((size_t)n * L_GRID + row) * C_DIM + c0 + lr]);
      s = red16(s);
      if (lr == 0) atomicAdd(&simpos[n * L_GRID + row], s);
    }
  }
}

// sim_neg = E @ Neg^T fused with S[r] += sum_m exp(sim_neg[r,m]).
// Each wave: 16(r) x 64(m) block = 4 accumulators; A fragment reused 4x.
// Cosine sims are in [-1,1] -> no max-subtraction needed for logsumexp.
__global__ void __launch_bounds__(128)
k_neg_expsum(const unsigned short* __restrict__ Ebf,
             const unsigned short* __restrict__ Neg,
             float* __restrict__ S) {
  const int lane = threadIdx.x & 31;
  const int tile = blockIdx.x * 4 + (threadIdx.x >> 5);  // 2048(r) * 64(mblk)
  const int m0 = (tile & 63) << 6;        // 64-col m block
  const int r0 = (tile >> 6) << 4;        // 16-row r tile
  const int lr = lane & 15;
  const int lh = lane >> 4;
  const unsigned short* Arow = Ebf + (size_t)(r0 + lr) * C_DIM + 8 * lh;
  const unsigned short* Brow = Neg + (size_t)(m0 + lr) * C_DIM + 8 * lh;

  v8f acc[4] = {};
#pragma unroll 2
  for (int c0 = 0; c0 < C_DIM; c0 += 32) {
    v16bf a = load_frag_a(Arow + c0);
#pragma unroll
    for (int t = 0; t < 4; ++t) {
      v16bf b = load_frag_b(Brow + (size_t)t * 16 * C_DIM + c0);
      acc[t] = wmma_bf16(a, b, acc[t]);
    }
  }
  // epilogue: all 4 m-blocks feed the same 16 rows -> sum exps, one atomic
#pragma unroll
  for (int j = 0; j < 8; ++j) {
    float s = __expf(acc[0][j]) + __expf(acc[1][j]) +
              __expf(acc[2][j]) + __expf(acc[3][j]);
    s = red16(s);
    if (lr == 0) atomicAdd(&S[r0 + j + 8 * lh], s);
  }
}

__global__ void k_final(const float* __restrict__ simpos,
                        const float* __restrict__ S,
                        float* __restrict__ out) {
  __shared__ float red[256];
  float acc = 0.f;
  for (int r = threadIdx.x; r < ROWS; r += 256) {
    const float sp = simpos[r];
    acc += logf(__expf(sp) + S[r]) - sp;   // logsumexp(row) - logit[0]
  }
  red[threadIdx.x] = acc;
  __syncthreads();
  for (int s = 128; s > 0; s >>= 1) {
    if (threadIdx.x < s) red[threadIdx.x] += red[threadIdx.x + s];
    __syncthreads();
  }
  if (threadIdx.x == 0) out[0] = red[0] / (float)ROWS;
}

// ---------------------------------------------------------------------------
extern "C" void kernel_launch(void* const* d_in, const int* in_sizes, int n_in,
                              void* d_out, int out_size, void* d_ws, size_t ws_size,
                              hipStream_t stream) {
  (void)in_sizes; (void)n_in; (void)out_size; (void)ws_size;
  const float* emb = (const float*)d_in[0];   // [32, 1024, 512] f32
  const float* wgt = (const float*)d_in[1];   // [1024, 1024] f32
  const int*   idx = (const int*)d_in[2];     // [4096] int32
  float* out = (float*)d_out;                 // scalar loss

  // workspace layout (~38.3 MiB -> whole working set lives in 192 MB L2)
  char* ws = (char*)d_ws;
  unsigned short* Ebf = (unsigned short*)ws;                                 // 32 MiB
  unsigned short* Wbf = (unsigned short*)(ws + (size_t)ROWS * C_DIM * 2);    // 2 MiB
  unsigned short* Neg = (unsigned short*)((char*)Wbf + (size_t)L_GRID * L_GRID * 2); // 4 MiB
  float* simpos = (float*)((char*)Neg + (size_t)M_NEG * C_DIM * 2);          // 128 KiB
  float* S      = simpos + ROWS;                                             // 128 KiB

  k_zero<<<(2 * ROWS + 255) / 256, 256, 0, stream>>>(simpos, 2 * ROWS);
  k_cvt_w<<<(L_GRID * L_GRID / 4) / 256, 256, 0, stream>>>(wgt, Wbf);
  k_normalize<<<ROWS / 8, 256, 0, stream>>>(emb, Ebf);
  k_gather<<<M_NEG / 8, 256, 0, stream>>>(Ebf, idx, Neg);
  k_target_simpos<<<(32 * 16 * N_BATCH) / 4, 128, 0, stream>>>(Wbf, Ebf, simpos);
  k_neg_expsum<<<((ROWS / 16) * (M_NEG / 64)) / 4, 128, 0, stream>>>(Ebf, Neg, S);
  k_final<<<1, 256, 0, stream>>>(simpos, S, out);
}
